// ScaledDotProductAttention_54846732370353
// MI455X (gfx1250) — matrix operand order repaired and checked by reading, not compiled, not run on hardware
//
#include <hip/hip_runtime.h>

typedef __attribute__((ext_vector_type(16))) __bf16 v16bf;
typedef __attribute__((ext_vector_type(8)))  float  v8f;

#define S_LEN 2048
#define D_DIM 64
#define QBLK  128   // queries per workgroup (8 waves x 16 rows)
#define KBLK  64    // keys per inner iteration
#define KSTRIDE 72  // Kt row stride (bf16): key-major [key][d], 16B-aligned rows
#define VSTRIDE 72  // Vt row stride (bf16): d-major [d][key], 16B-aligned rows

// Contraction index (even base) held by A-fragment VGPR-pair p (0..7) for a
// 16-bit 16x32 A matrix, contiguous order: lanes 0-15 hold K=0..15,
// lanes 16-31 hold K=16..31 (matches the Q^T B fragments).
__device__ __forceinline__ int a_kidx(int p, int half) {
  return 2 * p + 16 * half;
}

// Input s = (q.k)/16 (Q pre-scaled by 1/16). Returns sigmoid(q.k/8).
__device__ __forceinline__ float sigmoid_half_tanh(float s) {
#if __has_builtin(__builtin_amdgcn_tanhf)
  return __builtin_amdgcn_tanhf(s) * 0.5f + 0.5f;   // v_tanh_f32 (gfx1250)
#elif __has_builtin(__builtin_amdgcn_tanh_f32)
  return __builtin_amdgcn_tanh_f32(s) * 0.5f + 0.5f;
#else
  return __builtin_amdgcn_rcpf(1.0f + __expf(s * -2.0f));
#endif
}

__global__ __launch_bounds__(256, 2)
void sigmoid_attn_kernel(const float* __restrict__ Q,
                         const float* __restrict__ K,
                         const float* __restrict__ V,
                         float* __restrict__ O) {
  __shared__ __bf16 Kt[KBLK * KSTRIDE];   // [key][d]
  __shared__ __bf16 Vt[D_DIM * VSTRIDE];  // [d][key]  (transposed)

  const int bh   = blockIdx.y;            // b*H + h
  const int qblk = blockIdx.x * QBLK;
  const int tid  = threadIdx.x;
  const int wave = tid >> 5;
  const int lane = tid & 31;
  const int half = lane >> 4;
  const int l16  = lane & 15;

  const size_t head_off = (size_t)bh * S_LEN * D_DIM;
  const float* Qh = Q + head_off;
  const float* Kh = K + head_off;
  const float* Vh = V + head_off;
  float*       Oh = O + head_off;

  // Loader assignments:
  //  K: each thread owns 16 contiguous elems of one key row.
  const int kkey = tid >> 2;              // 0..63
  const int kcol = (tid & 3) * 16;        // 0,16,32,48
  //  V: each thread owns a 4x4 (key x d) block, transposed into Vt.
  const int vkey0 = (tid >> 4) * 4;       // 0,4,...,60
  const int vcol0 = (tid & 15) * 4;       // 0,4,...,60

  // ---- Q^T B-fragments (for S^T = K_tile * Q^T), loaded once per wave ----
  // B layout (32x16, 16-bit): lane holds column n = l16 (query),
  // contraction k = 16*half + e  ->  d = 32*frag + 16*half + e (contiguous).
  // Q pre-scaled by 1/16 so scores arrive as (q.k)/16.
  const int qrow = qblk + wave * 16 + l16;
  v16bf qb0, qb1;
  {
    const float* qp = Qh + (size_t)qrow * D_DIM + 16 * half;
#pragma unroll
    for (int e = 0; e < 16; ++e) {
      qb0[e] = (__bf16)(qp[e] * 0.0625f);
      qb1[e] = (__bf16)(qp[32 + e] * 0.0625f);
    }
  }

  v8f acc[4] = {v8f{}, v8f{}, v8f{}, v8f{}};

  for (int kb = 0; kb < S_LEN; kb += KBLK) {
    // Prefetch next key/value tile into caches (global_prefetch_b8).
    if (kb + KBLK < S_LEN) {
      __builtin_prefetch(Kh + (size_t)(kb + KBLK) * D_DIM + tid * 16, 0, 3);
      __builtin_prefetch(Vh + (size_t)(kb + KBLK) * D_DIM + tid * 16, 0, 3);
    }

    __syncthreads();  // everyone done reading previous LDS tile

    // ---- cooperative global -> LDS, f32 -> bf16 ----
    {
      // K: 16 contiguous elems -> 2x 16B LDS stores (rows 16B aligned)
      const float4* kgp =
          (const float4*)(Kh + (size_t)(kb + kkey) * D_DIM + kcol);
      union { __bf16 h[8]; uint4 u; } p0, p1;
      float4 f0 = kgp[0], f1 = kgp[1], f2 = kgp[2], f3 = kgp[3];
      p0.h[0] = (__bf16)f0.x; p0.h[1] = (__bf16)f0.y;
      p0.h[2] = (__bf16)f0.z; p0.h[3] = (__bf16)f0.w;
      p0.h[4] = (__bf16)f1.x; p0.h[5] = (__bf16)f1.y;
      p0.h[6] = (__bf16)f1.z; p0.h[7] = (__bf16)f1.w;
      p1.h[0] = (__bf16)f2.x; p1.h[1] = (__bf16)f2.y;
      p1.h[2] = (__bf16)f2.z; p1.h[3] = (__bf16)f2.w;
      p1.h[4] = (__bf16)f3.x; p1.h[5] = (__bf16)f3.y;
      p1.h[6] = (__bf16)f3.z; p1.h[7] = (__bf16)f3.w;
      uint4* kdst = (uint4*)&Kt[kkey * KSTRIDE + kcol];
      kdst[0] = p0.u;
      kdst[1] = p1.u;

      // V: 4x4 block transpose -> 4 packed ds_store_b64 (one per d column)
      const float* vgp = Vh + (size_t)(kb + vkey0) * D_DIM + vcol0;
      float4 r0 = *(const float4*)(vgp + 0 * D_DIM);
      float4 r1 = *(const float4*)(vgp + 1 * D_DIM);
      float4 r2 = *(const float4*)(vgp + 2 * D_DIM);
      float4 r3 = *(const float4*)(vgp + 3 * D_DIM);
      union { __bf16 h[4]; uint2 u; } c0, c1, c2, c3;
      c0.h[0] = (__bf16)r0.x; c0.h[1] = (__bf16)r1.x;
      c0.h[2] = (__bf16)r2.x; c0.h[3] = (__bf16)r3.x;
      c1.h[0] = (__bf16)r0.y; c1.h[1] = (__bf16)r1.y;
      c1.h[2] = (__bf16)r2.y; c1.h[3] = (__bf16)r3.y;
      c2.h[0] = (__bf16)r0.z; c2.h[1] = (__bf16)r1.z;
      c2.h[2] = (__bf16)r2.z; c2.h[3] = (__bf16)r3.z;
      c3.h[0] = (__bf16)r0.w; c3.h[1] = (__bf16)r1.w;
      c3.h[2] = (__bf16)r2.w; c3.h[3] = (__bf16)r3.w;
      *(uint2*)&Vt[(vcol0 + 0) * VSTRIDE + vkey0] = c0.u;
      *(uint2*)&Vt[(vcol0 + 1) * VSTRIDE + vkey0] = c1.u;
      *(uint2*)&Vt[(vcol0 + 2) * VSTRIDE + vkey0] = c2.u;
      *(uint2*)&Vt[(vcol0 + 3) * VSTRIDE + vkey0] = c3.u;
    }
    __syncthreads();

    // ---- GEMM1: S^T(64 keys x 16 queries) = K_tile(64x64) * Q^T(64x16) ----
    v8f st[4];
#pragma unroll
    for (int mt = 0; mt < 4; ++mt) {
      const __bf16* krow = &Kt[(mt * 16 + l16) * KSTRIDE];
      v16bf ka0, ka1;
#pragma unroll
      for (int p = 0; p < 8; ++p) {
        const int d = a_kidx(p, half);
        ka0[2 * p]     = krow[d];
        ka0[2 * p + 1] = krow[d + 1];
        ka1[2 * p]     = krow[32 + d];
        ka1[2 * p + 1] = krow[32 + d + 1];
      }
      v8f c = {};
      c = __builtin_amdgcn_wmma_f32_16x16x32_bf16(false, ka0, false, qb0,
                                                  (short)0, c, false, false);
      c = __builtin_amdgcn_wmma_f32_16x16x32_bf16(false, ka1, false, qb1,
                                                  (short)0, c, false, false);
      st[mt] = c;
    }

    // ---- sigmoid via hw tanh; S^T C-frags ARE the P A-frag layout ----
    v16bf pa0, pa1;  // P fragments for keys 0-31 and 32-63
#pragma unroll
    for (int e = 0; e < 8; ++e) {
      pa0[e]     = (__bf16)sigmoid_half_tanh(st[0][e]);
      pa0[e + 8] = (__bf16)sigmoid_half_tanh(st[1][e]);
      pa1[e]     = (__bf16)sigmoid_half_tanh(st[2][e]);
      pa1[e + 8] = (__bf16)sigmoid_half_tanh(st[3][e]);
    }

    // ---- GEMM2: O(16x64) += P(16x64) * V(64x64), 4 n-tiles x 2 k-steps ----
#pragma unroll
    for (int nt = 0; nt < 4; ++nt) {
      const __bf16* vrow = &Vt[(nt * 16 + l16) * VSTRIDE + 8 * half];
      v16bf vb0, vb1;  // B layout: lane col n = l16 (d), k = 8*half + e (e<8), 16 + 8*half + (e-8) (e>=8)
#pragma unroll
      for (int e = 0; e < 16; ++e) {
        vb0[e] = vrow[e < 8 ? e : e + 8];        // keys 0-31 slice
        vb1[e] = vrow[32 + (e < 8 ? e : e + 8)];   // keys 32-63 slice
      }
      acc[nt] = __builtin_amdgcn_wmma_f32_16x16x32_bf16(false, pa0, false, vb0,
                                                        (short)0, acc[nt], false, false);
      acc[nt] = __builtin_amdgcn_wmma_f32_16x16x32_bf16(false, pa1, false, vb1,
                                                        (short)0, acc[nt], false, false);
    }
  }

  // ---- store O: C layout lane n = d-col, regs = 8 query rows per half ----
#pragma unroll
  for (int nt = 0; nt < 4; ++nt) {
    const int d = nt * 16 + l16;
#pragma unroll
    for (int r = 0; r < 8; ++r) {
      const int q = qblk + wave * 16 + 8 * half + r;
      Oh[(size_t)q * D_DIM + d] = acc[nt][r];
    }
  }
}

extern "C" void kernel_launch(void* const* d_in, const int* in_sizes, int n_in,
                              void* d_out, int out_size, void* d_ws, size_t ws_size,
                              hipStream_t stream) {
  const float* q = (const float*)d_in[0];
  const float* k = (const float*)d_in[1];
  const float* v = (const float*)d_in[2];
  float* out = (float*)d_out;

  const int B = 4, H = 16;
  dim3 grid(S_LEN / QBLK, B * H);
  dim3 block(256);
  sigmoid_attn_kernel<<<grid, block, 0, stream>>>(q, k, v, out);
}
